// Attention_70781061038451
// MI455X (gfx1250) — matrix operand order repaired and checked by reading, not compiled, not run on hardware
//
#include <hip/hip_runtime.h>

// ---------- CDNA5 WMMA / TDM types ----------
typedef __attribute__((ext_vector_type(16))) __bf16 v16bf;
typedef __attribute__((ext_vector_type(8)))  __bf16 v8bf;
typedef __attribute__((ext_vector_type(8)))  float  v8f;
typedef __attribute__((ext_vector_type(4)))  unsigned int u32x4;
typedef __attribute__((ext_vector_type(8)))  int i32x8;
typedef __attribute__((ext_vector_type(4)))  int i32x4;

union F16U { v16bf v; v8bf h[2]; };

#define WMMA_BF16(a, b, c) \
  __builtin_amdgcn_wmma_f32_16x16x32_bf16(false, (a), false, (b), (short)0, (c), false, false)

#define S_LEN 2048
#define HID   4096
#define NH    32
#define HD    128

// ---------------------------------------------------------------
// TDM: async DMA of a 2D bf16 tile (rows x 128, row-major, contiguous
// rows of a (2048 x 128) tensor) from global memory into LDS.
// D# per CDNA5 ISA. Tracked by TENSORcnt.
// ---------------------------------------------------------------
__device__ __forceinline__ void tdm_load_rows_128(unsigned int lds_off,
                                                  const __bf16* gptr,
                                                  unsigned int rows) {
  unsigned long long ga = (unsigned long long)(uintptr_t)gptr;
  u32x4 g0;
  g0[0] = 1u;                                            // count=1, no gather
  g0[1] = lds_off;                                       // lds_addr (bytes)
  g0[2] = (unsigned int)(ga & 0xFFFFFFFFu);              // global_addr[31:0]
  g0[3] = (unsigned int)((ga >> 32) & 0x1FFFFFFu)        // global_addr[56:32]
          | (2u << 30);                                  // type=2 (image)
  i32x8 g1;
  g1[0] = (int)(1u << 16);        // workgroup_mask=0 | data_size=1 (2 bytes)
  g1[1] = (int)(128u << 16);      // atomic_barrier_addr=0 | tensor_dim0.lo16=128
  g1[2] = (int)(2048u << 16);     // tensor_dim0.hi16=0   | tensor_dim1.lo16=2048
  g1[3] = (int)(128u << 16);      // tensor_dim1.hi16=0   | tile_dim0=128
  g1[4] = (int)rows;              // tile_dim1=rows       | tile_dim2=0
  g1[5] = 128;                    // tensor_dim0_stride[31:0]=128
  g1[6] = 0;                      // stride0.hi16 | tensor_dim1_stride.lo16
  g1[7] = 0;
  i32x4 z4 = (i32x4){0, 0, 0, 0};            // groups 2/3 unused (2D tensor)
  i32x8 z8 = (i32x8){0, 0, 0, 0, 0, 0, 0, 0};
  __builtin_amdgcn_tensor_load_to_lds(g0, g1, z4, z4, z8, 0);
}

// ---------------------------------------------------------------
// Elementwise f32 -> bf16 convert
// ---------------------------------------------------------------
__global__ __launch_bounds__(256) void cvt_bf16(const float* __restrict__ src,
                                                __bf16* __restrict__ dst, int n) {
  int i = blockIdx.x * 256 + threadIdx.x;
  if (i < n) dst[i] = (__bf16)src[i];
}

// src is (N rows, K cols) f32; dst is (K rows, N cols) bf16  (i.e. W^T)
__global__ __launch_bounds__(256) void transpose_cvt(const float* __restrict__ src,
                                                     __bf16* __restrict__ dst,
                                                     int N, int K) {
  int i = blockIdx.x * 256 + threadIdx.x;
  if (i < N * K) {
    int n = i / K, k = i % K;
    dst[(size_t)k * N + n] = (__bf16)src[i];
  }
}

// ---------------------------------------------------------------
// Register-tiled bf16 WMMA GEMM:  C(MxN,f32) = A(MxK,bf16) * Bt(KxN,bf16)
// Wave tile 32x64: 2 M-tiles x 4 N-tiles, K stepped by 32.
// ---------------------------------------------------------------
__global__ __launch_bounds__(256) void gemm_bf16(const __bf16* __restrict__ A,
                                                 const __bf16* __restrict__ Bt,
                                                 float* __restrict__ C,
                                                 int M, int N, int K) {
  const int lane = threadIdx.x & 31;
  const int wave = blockIdx.x * (blockDim.x >> 5) + (threadIdx.x >> 5);
  const int ntiles = N >> 6;
  const int wm = wave / ntiles, wn = wave % ntiles;
  const int m0 = wm * 32, n0 = wn * 64;
  if (m0 >= M) return;
  const int ln = lane & 15, lh = lane >> 4;

  v8f acc[2][4];
#pragma unroll
  for (int i = 0; i < 2; ++i)
#pragma unroll
    for (int j = 0; j < 4; ++j) acc[i][j] = (v8f){0,0,0,0,0,0,0,0};

  for (int kb = 0; kb < K; kb += 32) {
    if (kb + 32 < K) {
      __builtin_prefetch(Bt + (size_t)(kb + 32 + lane) * N + n0, 0, 1);
      __builtin_prefetch(A + (size_t)(m0 + ln) * K + kb + 32, 0, 1);
    }
    v16bf a[2], b[4];
#pragma unroll
    for (int i = 0; i < 2; ++i) {
      const __bf16* p = A + (size_t)(m0 + i * 16 + ln) * K + kb + lh * 8;
      F16U t;
      t.h[0] = *(const v8bf*)p;
      t.h[1] = *(const v8bf*)(p + 16);
      a[i] = t.v;
    }
#pragma unroll
    for (int j = 0; j < 4; ++j)
#pragma unroll
      for (int s = 0; s < 16; ++s)
        b[j][s] = Bt[(size_t)(kb + (s & 7) + ((s >> 3) << 4) + lh * 8) * N + n0 + j * 16 + ln];
#pragma unroll
    for (int i = 0; i < 2; ++i)
#pragma unroll
      for (int j = 0; j < 4; ++j) acc[i][j] = WMMA_BF16(a[i], b[j], acc[i][j]);
  }
#pragma unroll
  for (int i = 0; i < 2; ++i)
#pragma unroll
    for (int j = 0; j < 4; ++j)
#pragma unroll
      for (int r = 0; r < 8; ++r)
        C[(size_t)(m0 + i * 16 + r + 8 * lh) * N + n0 + j * 16 + ln] = acc[i][j][r];
}

// ---------------------------------------------------------------
// RoPE on q; repack f32 (S, H) -> bf16 (head, S, HD)
// ---------------------------------------------------------------
__global__ __launch_bounds__(256) void rope_q(const float* __restrict__ q,
                                              const float* __restrict__ cs,
                                              const float* __restrict__ sn,
                                              __bf16* __restrict__ Qb) {
  int i = blockIdx.x * 256 + threadIdx.x;
  int s = i >> 12, c = i & 4095;
  int hh = c >> 7, d = c & 127;
  float qv = q[i];
  int base = (s << 12) + hh * HD;
  float rot = (d < 64) ? -q[base + d + 64] : q[base + d - 64];
  float o = qv * cs[(s << 7) + d] + rot * sn[(s << 7) + d];
  Qb[((size_t)hh * S_LEN + s) * HD + d] = (__bf16)o;
}

__global__ __launch_bounds__(256) void rope_kv(const float* __restrict__ k,
                                               const float* __restrict__ v,
                                               const float* __restrict__ cs,
                                               const float* __restrict__ sn,
                                               float* __restrict__ key_out,
                                               float* __restrict__ val_out,
                                               __bf16* __restrict__ Kb,
                                               __bf16* __restrict__ Vb) {
  int i = blockIdx.x * 256 + threadIdx.x;
  int s = i >> 7, d = i & 127;
  float kv = k[i];
  float rot = (d < 64) ? -k[(s << 7) + d + 64] : k[(s << 7) + d - 64];
  float kr = kv * cs[i] + rot * sn[i];
  key_out[i] = kr;
  Kb[i] = (__bf16)kr;
  float vv = v[i];
  val_out[i] = vv;
  Vb[i] = (__bf16)vv;
}

// ---------------------------------------------------------------
// Flash attention, S^T formulation (causal, 1 KV head / 32 q heads).
// Block = 256 thr (8 waves) handles 128 q rows of one head.
// Q^T staged once in LDS; K/V tiles double-buffered via TDM
// tensor_load_to_lds: tile i+1 streams in (TENSORcnt<=2 pipeline)
// while tile i is consumed by WMMAs.
// ---------------------------------------------------------------
__global__ __launch_bounds__(256) void flash_attn(const __bf16* __restrict__ Qb,
                                                  const __bf16* __restrict__ Kb,
                                                  const __bf16* __restrict__ Vb,
                                                  __bf16* __restrict__ attnb) {
  __shared__ __bf16 qt[HD * 128];        // Q^T tile: [d][qrel]      32 KB
  __shared__ __bf16 kraw[2][32 * HD];    // K tiles (TDM dest)     2x 8 KB
  __shared__ __bf16 vraw[2][32 * HD];    // V tiles (TDM dest)     2x 8 KB
  __shared__ __bf16 ps[8 * 16 * 32];     // per-wave P^T scratch      8 KB

  const int tid = threadIdx.x;
  const int lane = tid & 31, wvid = tid >> 5;
  const int ln = lane & 15, lh = lane >> 4;
  const int hh = blockIdx.y;
  const int qb0 = blockIdx.x * 128;
  const float inv_s = 0.08838834764831845f;  // 1/sqrt(128)

  const unsigned int lds_kb[2] = {(unsigned int)(uintptr_t)&kraw[0][0],
                                  (unsigned int)(uintptr_t)&kraw[1][0]};
  const unsigned int lds_vb[2] = {(unsigned int)(uintptr_t)&vraw[0][0],
                                  (unsigned int)(uintptr_t)&vraw[1][0]};
  const int nsteps = (qb0 + 128) / 32;

  // ---- Kick off tile 0 DMA immediately (overlaps Q^T staging) ----
  if (wvid == 0) {
    tdm_load_rows_128(lds_kb[0], Kb, 32u);
    tdm_load_rows_128(lds_vb[0], Vb, 32u);
  }

  // ---- Stage Q^T once: qt[d*128 + qrel] ----
  const __bf16* Qh = Qb + (size_t)hh * S_LEN * HD;
  for (int i = tid; i < 2048; i += 256) {
    int qrel = i >> 4, dg = (i & 15) << 3;
    v8bf qv = *(const v8bf*)(Qh + (size_t)(qb0 + qrel) * HD + dg);
#pragma unroll
    for (int e = 0; e < 8; ++e) qt[(dg + e) * 128 + qrel] = qv[e];
  }
  __syncthreads();

  // Loop-invariant Q^T B-fragments: lane -> q col (wvid*16+ln), 16 d rows in table order
  v16bf qf[4];
#pragma unroll
  for (int c = 0; c < 4; ++c)
#pragma unroll
    for (int s = 0; s < 16; ++s)
      qf[c][s] = qt[(c * 32 + (s & 7) + ((s >> 3) << 4) + lh * 8) * 128 + wvid * 16 + ln];

  v8f o[8];
#pragma unroll
  for (int dt = 0; dt < 8; ++dt) o[dt] = (v8f){0,0,0,0,0,0,0,0};
  float m_run = -1e30f, l_run = 0.f;  // per-lane stats for q column ln
  const int qrow = qb0 + wvid * 16 + ln;

  for (int step = 0; step < nsteps; ++step) {
    const int j0 = step * 32;
    const int cur = step & 1;
    // ---- Pipeline: issue tile step+1, wait for tile step ----
    if (wvid == 0) {
      if (step + 1 < nsteps) {
        tdm_load_rows_128(lds_kb[cur ^ 1], Kb + (size_t)(j0 + 32) * HD, 32u);
        tdm_load_rows_128(lds_vb[cur ^ 1], Vb + (size_t)(j0 + 32) * HD, 32u);
        __builtin_amdgcn_s_wait_tensorcnt(2);  // in-order: tile `step` landed
      } else {
        __builtin_amdgcn_s_wait_tensorcnt(0);
      }
    }
    __syncthreads();

    // ---- S^T = K_tile @ Q^T : two 16-kv tiles, K-dim 128 via 4 WMMAs ----
    const __bf16* kcur = &kraw[cur][0];
    v8f st[2];
#pragma unroll
    for (int t = 0; t < 2; ++t) {
      st[t] = (v8f){0,0,0,0,0,0,0,0};
#pragma unroll
      for (int c = 0; c < 4; ++c) {
        const __bf16* p = kcur + (t * 16 + ln) * HD + c * 32 + lh * 8;
        F16U a;
        a.h[0] = *(const v8bf*)p;
        a.h[1] = *(const v8bf*)(p + 16);
        st[t] = WMMA_BF16(a.v, qf[c], st[t]);
      }
    }

    // ---- Online softmax over kv for fixed q = ln ----
    // element (t, r, lane): kv = j0 + t*16 + r + 8*lh ; q = qrow
    float vals[16];
    float mx = -1e30f;
#pragma unroll
    for (int t = 0; t < 2; ++t)
#pragma unroll
      for (int r = 0; r < 8; ++r) {
        int kv = j0 + t * 16 + r + 8 * lh;
        float a = st[t][r] * inv_s;
        if (kv > qrow) a = -1e30f;
        vals[t * 8 + r] = a;
        mx = fmaxf(mx, a);
      }
    mx = fmaxf(mx, __shfl_xor(mx, 16, 32));
    float mnew = fmaxf(m_run, mx);

    __bf16* pw = &ps[wvid * 512];  // P^T scratch: [q=ln][kv 0..31]
    float sum = 0.f;
#pragma unroll
    for (int t = 0; t < 2; ++t)
#pragma unroll
      for (int r = 0; r < 8; ++r) {
        float p = __expf(vals[t * 8 + r] - mnew);
        sum += p;
        pw[ln * 32 + t * 16 + r + 8 * lh] = (__bf16)p;
      }
    sum += __shfl_xor(sum, 16, 32);
    float esc = __expf(m_run - mnew);
    l_run = l_run * esc + sum;
    m_run = mnew;

    // Rescale O: O row q = r + 8*lh -> fetch that q's esc via bpermute
#pragma unroll
    for (int r = 0; r < 8; ++r) {
      float er = __shfl(esc, r + 8 * lh, 32);
#pragma unroll
      for (int dt = 0; dt < 8; ++dt) o[dt][r] *= er;
    }

    // ---- O += P @ V ----
    const __bf16* vcur = &vraw[cur][0];
    F16U pf;
    pf.h[0] = *(const v8bf*)(pw + ln * 32 + lh * 8);
    pf.h[1] = *(const v8bf*)(pw + ln * 32 + 16 + lh * 8);
#pragma unroll
    for (int dt = 0; dt < 8; ++dt) {
      v16bf vf;
#pragma unroll
      for (int s = 0; s < 16; ++s)
        vf[s] = vcur[((s & 7) + ((s >> 3) << 4) + lh * 8) * HD + dt * 16 + ln];
      o[dt] = WMMA_BF16(pf.v, vf, o[dt]);
    }
    __syncthreads();  // all reads of buf[cur] done before TDM reuses it
  }

  // ---- Normalize and write attn bf16 (S, H), col = head*128 + d ----
#pragma unroll
  for (int r = 0; r < 8; ++r) {
    float lr = __shfl(l_run, r + 8 * lh, 32);
    float inv_l = 1.0f / lr;
    int srow = qb0 + wvid * 16 + r + 8 * lh;
#pragma unroll
    for (int dt = 0; dt < 8; ++dt)
      attnb[(size_t)srow * HID + hh * HD + dt * 16 + ln] = (__bf16)(o[dt][r] * inv_l);
  }
}

// ---------------------------------------------------------------
extern "C" void kernel_launch(void* const* d_in, const int* in_sizes, int n_in,
                              void* d_out, int out_size, void* d_ws, size_t ws_size,
                              hipStream_t stream) {
  (void)in_sizes; (void)n_in; (void)out_size; (void)ws_size;
  const float* hs  = (const float*)d_in[0];
  const float* cs  = (const float*)d_in[1];
  const float* sn  = (const float*)d_in[2];
  const float* wq  = (const float*)d_in[3];
  const float* wk  = (const float*)d_in[4];
  const float* wvw = (const float*)d_in[5];
  const float* dw  = (const float*)d_in[6];

  float* out     = (float*)d_out;
  float* key_out = out + (size_t)S_LEN * HID;
  float* val_out = key_out + (size_t)S_LEN * HD;

  char* w = (char*)d_ws;
  size_t off = 0;
  auto alloc = [&](size_t bytes) { void* p = w + off; off += (bytes + 255) & ~(size_t)255; return p; };

  __bf16* hsb   = (__bf16*)alloc((size_t)S_LEN * HID * 2);
  __bf16* wqT   = (__bf16*)alloc((size_t)HID * HID * 2);
  __bf16* wkT   = (__bf16*)alloc((size_t)HID * HD * 2);
  __bf16* wvT   = (__bf16*)alloc((size_t)HID * HD * 2);
  __bf16* dT    = (__bf16*)alloc((size_t)HID * HID * 2);
  float*  qf32  = (float*)alloc((size_t)S_LEN * HID * 4);
  float*  kf32  = (float*)alloc((size_t)S_LEN * HD * 4);
  float*  vf32  = (float*)alloc((size_t)S_LEN * HD * 4);
  __bf16* Qbb   = (__bf16*)alloc((size_t)NH * S_LEN * HD * 2);
  __bf16* Kbb   = (__bf16*)alloc((size_t)S_LEN * HD * 2);
  __bf16* Vbb   = (__bf16*)alloc((size_t)S_LEN * HD * 2);
  __bf16* attnb = (__bf16*)alloc((size_t)S_LEN * HID * 2);

  // 1) Precision / layout conversion
  cvt_bf16<<<(S_LEN * HID) / 256, 256, 0, stream>>>(hs, hsb, S_LEN * HID);
  transpose_cvt<<<(HID * HID) / 256, 256, 0, stream>>>(wq, wqT, HID, HID);
  transpose_cvt<<<(HD * HID) / 256, 256, 0, stream>>>(wk, wkT, HD, HID);
  transpose_cvt<<<(HD * HID) / 256, 256, 0, stream>>>(wvw, wvT, HD, HID);
  transpose_cvt<<<(HID * HID) / 256, 256, 0, stream>>>(dw, dT, HID, HID);

  // 2) Projections (WMMA GEMMs)
  gemm_bf16<<<(S_LEN / 32) * (HID / 64) / 8, 256, 0, stream>>>(hsb, wqT, qf32, S_LEN, HID, HID);
  gemm_bf16<<<(S_LEN / 32) * (HD / 64) / 8, 256, 0, stream>>>(hsb, wkT, kf32, S_LEN, HD, HID);
  gemm_bf16<<<(S_LEN / 32) * (HD / 64) / 8, 256, 0, stream>>>(hsb, wvT, vf32, S_LEN, HD, HID);

  // 3) RoPE + emit key/value outputs
  rope_q<<<(S_LEN * HID) / 256, 256, 0, stream>>>(qf32, cs, sn, Qbb);
  rope_kv<<<(S_LEN * HD) / 256, 256, 0, stream>>>(kf32, vf32, cs, sn, key_out, val_out, Kbb, Vbb);

  // 4) Flash attention (double-buffered TDM-staged K/V)
  flash_attn<<<dim3(S_LEN / 128, NH), 256, 0, stream>>>(Qbb, Kbb, Vbb, attnb);

  // 5) Output projection
  gemm_bf16<<<(S_LEN / 32) * (HID / 64) / 8, 256, 0, stream>>>(attnb, dT, out, S_LEN, HID, HID);
}
